// MPLModule_17471926960764
// MI455X (gfx1250) — matrix operand order repaired and checked by reading, not compiled, not run on hardware
//
#include <hip/hip_runtime.h>
#include <hip/hip_bf16.h>
#include <math.h>

// ---------------- constants ----------------
#define Bn   4096
#define Dn   1024
#define Cn   1000
#define Pn   8
#define NPROX (Cn*Pn)          // 8000
#define LEN_KEEP 870           // int(1024*0.85)
#define N_MASKED 154.0f
#define INV_TEMP 10.0f
#define MARGIN_F 0.3f

typedef __attribute__((ext_vector_type(8)))  __bf16 v8bf;
typedef __attribute__((ext_vector_type(16))) __bf16 v16bf;
typedef __attribute__((ext_vector_type(8)))  float  v8f;

// ---------------- helpers ----------------
__device__ inline unsigned short f2bf(float f) {
    unsigned int u = __float_as_uint(f);
    u = u + 0x7FFFu + ((u >> 16) & 1u);      // round-to-nearest-even
    return (unsigned short)(u >> 16);
}

__device__ inline float blockSum(float v, float* red) {
    int t = threadIdx.x;
    red[t] = v; __syncthreads();
    for (int s = 128; s > 0; s >>= 1) { if (t < s) red[t] += red[t + s]; __syncthreads(); }
    float r = red[0]; __syncthreads(); return r;
}
__device__ inline float blockMax(float v, float* red) {
    int t = threadIdx.x;
    red[t] = v; __syncthreads();
    for (int s = 128; s > 0; s >>= 1) { if (t < s) red[t] = fmaxf(red[t], red[t + s]); __syncthreads(); }
    float r = red[0]; __syncthreads(); return r;
}

// CDNA5 async global->LDS DMA (ASYNCcnt-tracked), 16B per lane
__device__ inline void async_cp16(unsigned lds_off, const void* gptr) {
    asm volatile("global_load_async_to_lds_b128 %0, %1, off"
                 :: "v"(lds_off), "v"(gptr) : "memory");
}
__device__ inline void wait_async0() {
    asm volatile("s_wait_asynccnt 0" ::: "memory");
}

// A fragment (16x32 bf16) from LDS tile (row stride 32 elems):
// lane l -> row (l&15); K = off..off+7, 16+off..16+off+7, off=8*(l>=16)
__device__ inline v16bf fragA_lds(const unsigned short* sA, int row, int lane) {
    const __bf16* p = (const __bf16*)(sA + row * 32 + ((lane >= 16) ? 8 : 0));
    union { v16bf v; v8bf h[2]; } u;
    u.h[0] = *(const v8bf*)p;
    u.h[1] = *(const v8bf*)(p + 16);
    return u.v;
}
// B fragment (32x16 bf16) from LDS tile: lane l -> column (l&15);
// K = off..off+7, 16+off..16+off+7, off=8*(l>=16)
__device__ inline v16bf fragB_lds(const unsigned short* sB, int col, int lane) {
    const __bf16* p = (const __bf16*)(sB + col * 32 + ((lane >= 16) ? 8 : 0));
    union { v16bf v; v8bf h[2]; } u;
    u.h[0] = *(const v8bf*)p;
    u.h[1] = *(const v8bf*)(p + 16);
    return u.v;
}

// Stage one K-step block tile into LDS: A rows [mblk,mblk+256), B rows [nblk,nblk+64), k [kk,kk+32)
__device__ inline void issue_tile(const unsigned short* __restrict__ A,
                                  const unsigned short* __restrict__ Bm,
                                  unsigned short* sA, unsigned short* sB,
                                  int mblk, int nblk, int kk, int wid, int lane) {
#pragma unroll
    for (int it = 0; it < 4; ++it) {               // A: 256 rows x 64B, 4 issues/wave
        int row = wid * 32 + it * 8 + (lane >> 2);
        int chunk = lane & 3;
        const void* g = (const void*)(A + (size_t)(mblk + row) * Dn + kk + chunk * 8);
        unsigned loff = (unsigned)(uintptr_t)(sA + row * 32 + chunk * 8);
        async_cp16(loff, g);
    }
    {                                              // B: 64 rows x 64B, 1 issue/wave
        int i = wid * 32 + lane;
        int row = i >> 2, chunk = i & 3;
        const void* g = (const void*)(Bm + (size_t)(nblk + row) * Dn + kk + chunk * 8);
        unsigned loff = (unsigned)(uintptr_t)(sB + row * 32 + chunk * 8);
        async_cp16(loff, g);
    }
}

// ---------------- prep kernels ----------------
__global__ void k_zero_acc(float* acc) {
    if (threadIdx.x < 2) acc[threadIdx.x] = 0.0f;
}

// per-row: feature L2-norm -> featn bf16; rank-count noise -> mask; masked feats bf16
__global__ void k_prep_row(const float* __restrict__ feat, const float* __restrict__ noise,
                           unsigned short* __restrict__ featn, unsigned short* __restrict__ maskedbf,
                           unsigned char* __restrict__ mask_u8) {
    __shared__ float sf[Dn];
    __shared__ float sn[Dn];
    __shared__ float red[256];
    int b = blockIdx.x, t = threadIdx.x;
    float ss = 0.0f;
    for (int j = t; j < Dn; j += 256) {
        float f = feat[b * Dn + j];
        sf[j] = f; sn[j] = noise[b * Dn + j];
        ss += f * f;
    }
    __syncthreads();
    float nrm = sqrtf(blockSum(ss, red));
    float inv = 1.0f / fmaxf(nrm, 1e-12f);

    int j0 = t * 4;
    float nj0 = sn[j0], nj1 = sn[j0 + 1], nj2 = sn[j0 + 2], nj3 = sn[j0 + 3];
    int c0 = 0, c1 = 0, c2 = 0, c3 = 0;
    for (int k = 0; k < Dn; ++k) {
        float nk = sn[k];
        c0 += (nk < nj0) || (nk == nj0 && k < j0);
        c1 += (nk < nj1) || (nk == nj1 && k < j0 + 1);
        c2 += (nk < nj2) || (nk == nj2 && k < j0 + 2);
        c3 += (nk < nj3) || (nk == nj3 && k < j0 + 3);
    }
    int cnt[4] = {c0, c1, c2, c3};
#pragma unroll
    for (int q = 0; q < 4; ++q) {
        int j = j0 + q;
        int kept = cnt[q] < LEN_KEEP;
        mask_u8[b * Dn + j] = (unsigned char)kept;
        float f = sf[j];
        featn[b * Dn + j] = f2bf(f * inv);
        maskedbf[b * Dn + j] = f2bf(kept ? f : 0.0f);
    }
}

// per-proxy-row: renormalize + convert to bf16
__global__ void k_prep_prox(const float* __restrict__ prox, unsigned short* __restrict__ proxbf) {
    __shared__ float red[256];
    int r = blockIdx.x, t = threadIdx.x;
    float v[4]; float ss = 0.0f;
#pragma unroll
    for (int q = 0; q < 4; ++q) { v[q] = prox[r * Dn + t + 256 * q]; ss += v[q] * v[q]; }
    float nrm = sqrtf(blockSum(ss, red));
    float inv = 1.0f / fmaxf(nrm, 1e-12f);
#pragma unroll
    for (int q = 0; q < 4; ++q) proxbf[r * Dn + t + 256 * q] = f2bf(v[q] * inv);
}

__global__ void k_f2bf(const float* __restrict__ s, unsigned short* __restrict__ d, int n) {
    for (int i = blockIdx.x * blockDim.x + threadIdx.x; i < n; i += gridDim.x * blockDim.x)
        d[i] = f2bf(s[i]);
}

// ---------------- WMMA GEMMs (LDS double-buffered via async DMA) ----------------
// sim GEMM: [4096,1024] x [1024,8000], fused max-over-8-proxies -> class_sim[4096,1000]
// grid: 16 m-blocks x 125 n-blocks; block tile 256x64; wave (of 8): 64x32
__global__ void __launch_bounds__(256) k_sim(const unsigned short* __restrict__ A,
                                             const unsigned short* __restrict__ Pm,
                                             float* __restrict__ class_sim) {
    __shared__ __align__(16) unsigned short sA[2][256 * 32];
    __shared__ __align__(16) unsigned short sB[2][64 * 32];
    int t = threadIdx.x, lane = t & 31, wid = t >> 5;
    int mb = blockIdx.x & 15, nb = blockIdx.x >> 4;
    int mblk = mb * 256, nblk = nb * 64;
    int mw = wid >> 1, nw = wid & 1;      // wave rows: mw*64..+63, cols: nw*32..+31

    v8f acc[4][2];
#pragma unroll
    for (int i = 0; i < 4; ++i)
#pragma unroll
        for (int j = 0; j < 2; ++j)
#pragma unroll
            for (int e = 0; e < 8; ++e) acc[i][j][e] = 0.0f;

    issue_tile(A, Pm, sA[0], sB[0], mblk, nblk, 0, wid, lane);
    wait_async0();
    __syncthreads();

    for (int ks = 0; ks < 32; ++ks) {
        const unsigned short* cA = sA[ks & 1];
        const unsigned short* cB = sB[ks & 1];
        if (ks + 1 < 32)
            issue_tile(A, Pm, sA[(ks + 1) & 1], sB[(ks + 1) & 1], mblk, nblk, (ks + 1) * 32, wid, lane);
        if (ks + 2 < 32) {  // GL2 prefetch of tile k+2 while DMA k+1 is in flight
            __builtin_prefetch((const void*)(A + (size_t)(mblk + wid * 32 + (lane >> 2)) * Dn + (ks + 2) * 32), 0, 0);
            __builtin_prefetch((const void*)(Pm + (size_t)(nblk + (lane >> 2)) * Dn + (ks + 2) * 32), 0, 0);
        }
        v16bf a[4], b[2];
#pragma unroll
        for (int i = 0; i < 4; ++i) a[i] = fragA_lds(cA, mw * 64 + i * 16 + (lane & 15), lane);
#pragma unroll
        for (int j = 0; j < 2; ++j) b[j] = fragB_lds(cB, nw * 32 + j * 16 + (lane & 15), lane);
#pragma unroll
        for (int i = 0; i < 4; ++i)
#pragma unroll
            for (int j = 0; j < 2; ++j)
                acc[i][j] = __builtin_amdgcn_wmma_f32_16x16x32_bf16(false, a[i], false, b[j],
                                                                    (short)0, acc[i][j], false, false);
        wait_async0();
        __syncthreads();
    }

    // epilogue: max over groups of 8 proxy columns -> class_sim
#pragma unroll
    for (int i = 0; i < 4; ++i) {
#pragma unroll
        for (int j = 0; j < 2; ++j) {
            int clsbase = (nblk + nw * 32 + j * 16) >> 3;
#pragma unroll
            for (int e = 0; e < 8; ++e) {
                float v = acc[i][j][e];
                v = fmaxf(v, __shfl_xor(v, 1));
                v = fmaxf(v, __shfl_xor(v, 2));
                v = fmaxf(v, __shfl_xor(v, 4));
                if ((lane & 7) == 0) {
                    int cls = clsbase + ((lane >> 3) & 1);
                    int row = mblk + mw * 64 + i * 16 + ((lane >= 16) ? 8 : 0) + e;
                    class_sim[row * Cn + cls] = v;
                }
            }
        }
    }
}

// generic [4096,1024] x [1024,1024](row-major W, contracted over last dim) + bias -> f32
// grid: 16 m-blocks x 16 n-blocks; block tile 256x64; wave: 64x32
__global__ void __launch_bounds__(256) k_gemm_bias(const unsigned short* __restrict__ A,
                                                   const unsigned short* __restrict__ W,
                                                   const float* __restrict__ bias,
                                                   float* __restrict__ out) {
    __shared__ __align__(16) unsigned short sA[2][256 * 32];
    __shared__ __align__(16) unsigned short sB[2][64 * 32];
    int t = threadIdx.x, lane = t & 31, wid = t >> 5;
    int mb = blockIdx.x & 15, nb = blockIdx.x >> 4;
    int mblk = mb * 256, nblk = nb * 64;
    int mw = wid >> 1, nw = wid & 1;

    v8f acc[4][2];
#pragma unroll
    for (int i = 0; i < 4; ++i)
#pragma unroll
        for (int j = 0; j < 2; ++j)
#pragma unroll
            for (int e = 0; e < 8; ++e) acc[i][j][e] = 0.0f;

    issue_tile(A, W, sA[0], sB[0], mblk, nblk, 0, wid, lane);
    wait_async0();
    __syncthreads();

    for (int ks = 0; ks < 32; ++ks) {
        const unsigned short* cA = sA[ks & 1];
        const unsigned short* cB = sB[ks & 1];
        if (ks + 1 < 32)
            issue_tile(A, W, sA[(ks + 1) & 1], sB[(ks + 1) & 1], mblk, nblk, (ks + 1) * 32, wid, lane);
        if (ks + 2 < 32) {
            __builtin_prefetch((const void*)(A + (size_t)(mblk + wid * 32 + (lane >> 2)) * Dn + (ks + 2) * 32), 0, 0);
            __builtin_prefetch((const void*)(W + (size_t)(nblk + (lane >> 2)) * Dn + (ks + 2) * 32), 0, 0);
        }
        v16bf a[4], b[2];
#pragma unroll
        for (int i = 0; i < 4; ++i) a[i] = fragA_lds(cA, mw * 64 + i * 16 + (lane & 15), lane);
#pragma unroll
        for (int j = 0; j < 2; ++j) b[j] = fragB_lds(cB, nw * 32 + j * 16 + (lane & 15), lane);
#pragma unroll
        for (int i = 0; i < 4; ++i)
#pragma unroll
            for (int j = 0; j < 2; ++j)
                acc[i][j] = __builtin_amdgcn_wmma_f32_16x16x32_bf16(false, a[i], false, b[j],
                                                                    (short)0, acc[i][j], false, false);
        wait_async0();
        __syncthreads();
    }

#pragma unroll
    for (int i = 0; i < 4; ++i) {
        int rbase = mblk + mw * 64 + i * 16 + ((lane >= 16) ? 8 : 0);
#pragma unroll
        for (int j = 0; j < 2; ++j) {
            int ncol = nblk + nw * 32 + j * 16 + (lane & 15);
            float bia = bias[ncol];
#pragma unroll
            for (int e = 0; e < 8; ++e)
                out[(size_t)(rbase + e) * Dn + ncol] = acc[i][j][e] + bia;
        }
    }
}

// ---------------- per-row post kernels ----------------
__global__ void k_proxy_loss(const float* __restrict__ class_sim, const int* __restrict__ labels,
                             float* __restrict__ accum) {
    __shared__ float red[256];
    int b = blockIdx.x, t = threadIdx.x;
    int lbl = labels[b];
    const float* row = class_sim + b * Cn;
    float mx = -1e30f;
    for (int j = t; j < Cn; j += 256) {
        float lg = (row[j] - ((j == lbl) ? MARGIN_F : 0.0f)) * INV_TEMP;
        mx = fmaxf(mx, lg);
    }
    mx = blockMax(mx, red);
    float se = 0.0f;
    for (int j = t; j < Cn; j += 256) {
        float lg = (row[j] - ((j == lbl) ? MARGIN_F : 0.0f)) * INV_TEMP;
        se += expf(lg - mx);
    }
    se = blockSum(se, red);
    if (t == 0) {
        float lg_l = (row[lbl] - MARGIN_F) * INV_TEMP;
        float loss = -(lg_l - mx - logf(se));
        atomicAdd(accum, loss * (1.0f / (float)Bn));
    }
}

// LN+ReLU on enc_raw, proxy attention over the label's 8 proxies, combined -> bf16
__global__ void k_enc_post(const float* __restrict__ enc_raw, const float* __restrict__ g,
                           const float* __restrict__ beta, const float* __restrict__ proxies,
                           const int* __restrict__ labels, unsigned short* __restrict__ comb_bf) {
    __shared__ float red[256];
    __shared__ float sdots[Pn];
    __shared__ float satt[Pn];
    int b = blockIdx.x, t = threadIdx.x;
    float x[4]; float s = 0.0f, s2 = 0.0f;
#pragma unroll
    for (int q = 0; q < 4; ++q) {
        int j = t + 256 * q;
        x[q] = enc_raw[b * Dn + j];
        s += x[q]; s2 += x[q] * x[q];
    }
    s = blockSum(s, red); s2 = blockSum(s2, red);
    float mu = s * (1.0f / Dn);
    float var = s2 * (1.0f / Dn) - mu * mu;
    float rs = rsqrtf(var + 1e-5f);
    if (t < Pn) sdots[t] = 0.0f;
    __syncthreads();

    const float* pl = proxies + (size_t)labels[b] * Pn * Dn;
    float part[Pn];
#pragma unroll
    for (int p = 0; p < Pn; ++p) part[p] = 0.0f;
#pragma unroll
    for (int q = 0; q < 4; ++q) {
        int j = t + 256 * q;
        float e = fmaxf((x[q] - mu) * rs * g[j] + beta[j], 0.0f);
#pragma unroll
        for (int p = 0; p < Pn; ++p) part[p] += e * pl[p * Dn + j];
    }
#pragma unroll
    for (int p = 0; p < Pn; ++p) atomicAdd(&sdots[p], part[p]);
    __syncthreads();
    if (t == 0) {
        float m = -1e30f;
#pragma unroll
        for (int p = 0; p < Pn; ++p) m = fmaxf(m, sdots[p] * INV_TEMP);
        float ssum = 0.0f; float ev[Pn];
#pragma unroll
        for (int p = 0; p < Pn; ++p) { ev[p] = expf(sdots[p] * INV_TEMP - m); ssum += ev[p]; }
#pragma unroll
        for (int p = 0; p < Pn; ++p) satt[p] = ev[p] / ssum;
    }
    __syncthreads();
#pragma unroll
    for (int q = 0; q < 4; ++q) {
        int j = t + 256 * q;
        float c = 0.0f;
#pragma unroll
        for (int p = 0; p < Pn; ++p) c += satt[p] * pl[p * Dn + j];
        comb_bf[b * Dn + j] = f2bf(c);
    }
}

// LN on dec_raw, masked MSE vs features -> accumulate recon loss
__global__ void k_dec_post(const float* __restrict__ dec_raw, const float* __restrict__ g,
                           const float* __restrict__ beta, const float* __restrict__ feat,
                           const unsigned char* __restrict__ mask_u8, float* __restrict__ accum) {
    __shared__ float red[256];
    int b = blockIdx.x, t = threadIdx.x;
    float x[4]; float s = 0.0f, s2 = 0.0f;
#pragma unroll
    for (int q = 0; q < 4; ++q) {
        int j = t + 256 * q;
        x[q] = dec_raw[b * Dn + j];
        s += x[q]; s2 += x[q] * x[q];
    }
    s = blockSum(s, red); s2 = blockSum(s2, red);
    float mu = s * (1.0f / Dn);
    float var = s2 * (1.0f / Dn) - mu * mu;
    float rs = rsqrtf(var + 1e-5f);
    float ps = 0.0f;
#pragma unroll
    for (int q = 0; q < 4; ++q) {
        int j = t + 256 * q;
        float r = (x[q] - mu) * rs * g[j] + beta[j];
        float diff = r - feat[b * Dn + j];
        ps += diff * diff * (1.0f - (float)mask_u8[b * Dn + j]);
    }
    ps = blockSum(ps, red);
    if (t == 0) atomicAdd(accum + 1, ps / (N_MASKED * (float)Bn));
}

__global__ void k_final(const float* __restrict__ acc, float* __restrict__ out) {
    if (threadIdx.x == 0) out[0] = acc[0] + acc[1];
}

// ---------------- launcher ----------------
extern "C" void kernel_launch(void* const* d_in, const int* in_sizes, int n_in,
                              void* d_out, int out_size, void* d_ws, size_t ws_size,
                              hipStream_t stream) {
    const float* features = (const float*)d_in[0];
    const float* proxies  = (const float*)d_in[1];
    const float* noise    = (const float*)d_in[2];
    const float* enc_w    = (const float*)d_in[3];
    const float* enc_b    = (const float*)d_in[4];
    const float* enc_g    = (const float*)d_in[5];
    const float* enc_beta = (const float*)d_in[6];
    const float* dec_w    = (const float*)d_in[7];
    const float* dec_b    = (const float*)d_in[8];
    const float* dec_g    = (const float*)d_in[9];
    const float* dec_beta = (const float*)d_in[10];
    const int*   labels   = (const int*)d_in[11];
    float* out = (float*)d_out;

    char* ws = (char*)d_ws;
    size_t off = 0;
    auto carve = [&](size_t bytes) -> void* {
        off = (off + 255) & ~(size_t)255;
        void* p = ws + off;
        off += bytes;
        return p;
    };
    unsigned short* featn    = (unsigned short*)carve((size_t)Bn * Dn * 2);
    unsigned short* proxbf   = (unsigned short*)carve((size_t)NPROX * Dn * 2);
    unsigned short* maskedbf = (unsigned short*)carve((size_t)Bn * Dn * 2);
    unsigned char*  mask_u8  = (unsigned char*)carve((size_t)Bn * Dn);
    unsigned short* encw_bf  = (unsigned short*)carve((size_t)Dn * Dn * 2);
    unsigned short* decw_bf  = (unsigned short*)carve((size_t)Dn * Dn * 2);
    float*          class_sim= (float*)carve((size_t)Bn * Cn * 4);
    float*          enc_raw  = (float*)carve((size_t)Bn * Dn * 4);
    unsigned short* comb_bf  = (unsigned short*)carve((size_t)Bn * Dn * 2);
    float*          dec_raw  = (float*)carve((size_t)Bn * Dn * 4);
    float*          acc      = (float*)carve(2 * 4);

    k_zero_acc<<<1, 32, 0, stream>>>(acc);
    k_prep_row<<<Bn, 256, 0, stream>>>(features, noise, featn, maskedbf, mask_u8);
    k_prep_prox<<<NPROX, 256, 0, stream>>>(proxies, proxbf);
    k_f2bf<<<1024, 256, 0, stream>>>(enc_w, encw_bf, Dn * Dn);
    k_f2bf<<<1024, 256, 0, stream>>>(dec_w, decw_bf, Dn * Dn);

    // sim GEMM: 16 m-blocks x 125 n-blocks (block tile 256x64)
    k_sim<<<16 * 125, 256, 0, stream>>>(featn, proxbf, class_sim);
    k_proxy_loss<<<Bn, 256, 0, stream>>>(class_sim, labels, acc);

    // encoder GEMM: 16 x 16 blocks
    k_gemm_bias<<<16 * 16, 256, 0, stream>>>(maskedbf, encw_bf, enc_b, enc_raw);
    k_enc_post<<<Bn, 256, 0, stream>>>(enc_raw, enc_g, enc_beta, proxies, labels, comb_bf);

    // decoder GEMM
    k_gemm_bias<<<16 * 16, 256, 0, stream>>>(comb_bf, decw_bf, dec_b, dec_raw);
    k_dec_post<<<Bn, 256, 0, stream>>>(dec_raw, dec_g, dec_beta, features, mask_u8, acc);

    k_final<<<1, 32, 0, stream>>>(acc, out);
}
